// AttentionEncoderModel_47742856462555
// MI455X (gfx1250) — matrix operand order repaired and checked by reading, not compiled, not run on hardware
//
#include <hip/hip_runtime.h>
#include <math.h>

typedef __bf16 bf16;
typedef __attribute__((ext_vector_type(16))) __bf16        v16bf;
typedef __attribute__((ext_vector_type(8)))  float         v8f;
typedef __attribute__((ext_vector_type(4)))  unsigned int  u32x4;

#define NT 256

union FragA { v16bf v; unsigned int u[8]; };
union FragC { v8f   v; float        f[8]; };

// ---------------------------------------------------------------------------
// gfx1250 async global -> LDS copy (ASYNCcnt path).  dsaddr = low 32 bits of
// the generic shared pointer (flat LDS aperture truncates to the byte offset).
// ---------------------------------------------------------------------------
__device__ __forceinline__ void async_load_b128(bf16* lds_dst, const bf16* g) {
  const unsigned dst = (unsigned)(size_t)lds_dst;
  const unsigned long long ga = (unsigned long long)(size_t)g;
  asm volatile("global_load_async_to_lds_b128 %0, %1, off"
               :: "v"(dst), "v"(ga) : "memory");
}
__device__ __forceinline__ void wait_async() {
  asm volatile("s_wait_asynccnt 0" ::: "memory");
}

// ---------------------------------------------------------------------------
// WMMA fragment loaders (16x16x32 bf16, wave32; contiguous k order used for both A and B).
// A (16x32): lanes 0-15 M=lane, K 0..15; lanes 16-31 M=lane-16, K 16..31
//   -> 16 contiguous elements: two 16B loads at p and p+8 elements.
// B (32x16, source stored as B[n][k]): lanes 0-15 N=lane, K 0..15; lanes 16-31 K 16..31
//   -> 16 contiguous elements: two 16B loads at p and p+8 elements.
// Strides must keep 16B alignment (stride*2 % 16 == 0).
// ---------------------------------------------------------------------------
__device__ __forceinline__ void load_fragA(FragA& f, const bf16* base, int stride,
                                           int rowBase, int lane) {
  const int row  = rowBase + (lane & 15);
  const int koff = (lane >> 4) << 4;            // 0 or 16
  const bf16* p = base + row * stride + koff;
  *reinterpret_cast<u32x4*>(&f.u[0]) = *reinterpret_cast<const u32x4*>(p);
  *reinterpret_cast<u32x4*>(&f.u[4]) = *reinterpret_cast<const u32x4*>(p + 8);
}

__device__ __forceinline__ void load_fragB(FragA& f, const bf16* base, int stride,
                                           int colBase, int lane) {
  const int col  = colBase + (lane & 15);
  const int koff = (lane >> 4) << 4;            // 0 or 16
  const bf16* p = base + col * stride + koff;
  *reinterpret_cast<u32x4*>(&f.u[0]) = *reinterpret_cast<const u32x4*>(p);
  *reinterpret_cast<u32x4*>(&f.u[4]) = *reinterpret_cast<const u32x4*>(p + 8);
}

// ---------------------------------------------------------------------------
// Generic bf16 WMMA GEMM:  C[M,N] = act(A[M,K] @ W[K,N] + bias) (+pos)
// Block tile 128x128x32, 8 waves (2 M x 4 N), wave tile 64x32 (4x2 frags).
// A tile staged via async global->LDS b128; B tile transposed into LDS.
// ACT: 0=none 1=relu 2=tanh 3=gelu(exact)
// ---------------------------------------------------------------------------
template<int ACT, bool ADD_POS>
__global__ void __launch_bounds__(NT)
gemm_bf16_kernel(const bf16* __restrict__ A, const bf16* __restrict__ W,
                 const float* __restrict__ bias, const float* __restrict__ pos,
                 bf16* __restrict__ Cbf, float* __restrict__ Cf,
                 int M, int N, int K)
{
  constexpr int BM = 128, BN = 128, BK = 32, LDT = 40;  // 80B rows: 16B-aligned, bank-safe
  __shared__ bf16 As[BM * LDT];
  __shared__ bf16 Bs[BN * LDT];          // stored transposed: Bs[n][k]

  const int tid  = threadIdx.x;
  const int lane = tid & 31;
  const int wid  = tid >> 5;
  const int m0   = blockIdx.y * BM;
  const int n0   = blockIdx.x * BN;
  const int wm   = (wid & 1) * 64;
  const int wn   = (wid >> 1) * 32;

  FragC acc[4][2];
#pragma unroll
  for (int i = 0; i < 4; ++i)
#pragma unroll
    for (int j = 0; j < 2; ++j)
#pragma unroll
      for (int r = 0; r < 8; ++r) acc[i][j].f[r] = 0.0f;

  for (int k0 = 0; k0 < K; k0 += BK) {
    // ---- A tile: 512 x 16B async chunks (uniform EXEC) ----
    {
      const bf16* Ab = A + (size_t)m0 * K + k0;
#pragma unroll
      for (int c = tid; c < (BM * BK) / 8; c += NT) {
        const int m = c >> 2, kc = (c & 3) << 3;
        async_load_b128(&As[m * LDT + kc], Ab + (size_t)m * K + kc);
      }
    }
    // ---- B tile: b128 global read, transposed scalar LDS stores ----
    for (int c = tid; c < (BK * BN) / 8; c += NT) {
      const int k = c >> 4, nc = (c & 15) << 3;
      const u32x4 w = *reinterpret_cast<const u32x4*>(&W[(size_t)(k0 + k) * N + (n0 + nc)]);
      const bf16* wp = reinterpret_cast<const bf16*>(&w);
#pragma unroll
      for (int j = 0; j < 8; ++j) Bs[(nc + j) * LDT + k] = wp[j];
    }
    wait_async();
    __syncthreads();

    FragA afr[4], bfr[2];
#pragma unroll
    for (int sm = 0; sm < 4; ++sm) load_fragA(afr[sm], As, LDT, wm + sm * 16, lane);
#pragma unroll
    for (int sn = 0; sn < 2; ++sn) load_fragB(bfr[sn], Bs, LDT, wn + sn * 16, lane);

#pragma unroll
    for (int sm = 0; sm < 4; ++sm)
#pragma unroll
      for (int sn = 0; sn < 2; ++sn)
        acc[sm][sn].v = __builtin_amdgcn_wmma_f32_16x16x32_bf16(
            false, afr[sm].v, false, bfr[sn].v, (short)0, acc[sm][sn].v, false, false);
    __syncthreads();
  }

  const int rofs = (lane >> 4) * 8;
  const int nl   = lane & 15;
#pragma unroll
  for (int sm = 0; sm < 4; ++sm) {
#pragma unroll
    for (int sn = 0; sn < 2; ++sn) {
      const int n = n0 + wn + sn * 16 + nl;
      const float bv = bias[n];
#pragma unroll
      for (int r = 0; r < 8; ++r) {
        const int m = m0 + wm + sm * 16 + rofs + r;
        float v = acc[sm][sn].f[r] + bv;
        if (ACT == 1)      v = fmaxf(v, 0.0f);
        else if (ACT == 2) v = tanhf(v);
        else if (ACT == 3) v = 0.5f * v * (1.0f + erff(v * 0.70710678118654752f));
        if (ADD_POS)       v += pos[(size_t)(m & 511) * 256 + n];  // S=512, D=256
        const size_t idx = (size_t)m * N + n;
        if (Cbf) Cbf[idx] = (bf16)v;
        if (Cf)  Cf[idx]  = v;
      }
    }
  }
}

// ---------------------------------------------------------------------------
// LayerNorm over D=256 (one row per block of 256 threads)
// ---------------------------------------------------------------------------
__global__ void __launch_bounds__(256)
layernorm_kernel(const float* __restrict__ X, const float* __restrict__ g,
                 const float* __restrict__ bta, bf16* __restrict__ Obf)
{
  const int row = blockIdx.x, tid = threadIdx.x;
  const size_t idx = (size_t)row * 256 + tid;
  const float v = X[idx];
  float s = v, s2 = v * v;
#pragma unroll
  for (int off = 16; off; off >>= 1) { s += __shfl_xor(s, off, 32); s2 += __shfl_xor(s2, off, 32); }
  __shared__ float rs[8], rs2[8];
  if ((tid & 31) == 0) { rs[tid >> 5] = s; rs2[tid >> 5] = s2; }
  __syncthreads();
  s = 0.0f; s2 = 0.0f;
#pragma unroll
  for (int i = 0; i < 8; ++i) { s += rs[i]; s2 += rs2[i]; }
  const float mu  = s * (1.0f / 256.0f);
  const float var = s2 * (1.0f / 256.0f) - mu * mu;
  const float inv = rsqrtf(var + 1e-5f);
  Obf[idx] = (bf16)((v - mu) * inv * g[tid] + bta[tid]);
}

// x = LN(x + a); writes x (fp32, in place) and bf16 copy
__global__ void __launch_bounds__(256)
add_layernorm_kernel(float* __restrict__ X, const float* __restrict__ Aadd,
                     const float* __restrict__ g, const float* __restrict__ bta,
                     bf16* __restrict__ Obf)
{
  const int row = blockIdx.x, tid = threadIdx.x;
  const size_t idx = (size_t)row * 256 + tid;
  const float v = X[idx] + Aadd[idx];
  float s = v, s2 = v * v;
#pragma unroll
  for (int off = 16; off; off >>= 1) { s += __shfl_xor(s, off, 32); s2 += __shfl_xor(s2, off, 32); }
  __shared__ float rs[8], rs2[8];
  if ((tid & 31) == 0) { rs[tid >> 5] = s; rs2[tid >> 5] = s2; }
  __syncthreads();
  s = 0.0f; s2 = 0.0f;
#pragma unroll
  for (int i = 0; i < 8; ++i) { s += rs[i]; s2 += rs2[i]; }
  const float mu  = s * (1.0f / 256.0f);
  const float var = s2 * (1.0f / 256.0f) - mu * mu;
  const float inv = rsqrtf(var + 1e-5f);
  const float o = (v - mu) * inv * g[tid] + bta[tid];
  X[idx]   = o;
  Obf[idx] = (bf16)o;
}

__global__ void add_bf_kernel(float* __restrict__ X, const float* __restrict__ Hd,
                              bf16* __restrict__ Xbf, size_t n)
{
  const size_t stride = (size_t)gridDim.x * blockDim.x;
  for (size_t i = (size_t)blockIdx.x * blockDim.x + threadIdx.x; i < n; i += stride) {
    const float v = X[i] + Hd[i];
    X[i] = v;
    Xbf[i] = (bf16)v;
  }
}

__global__ void f2bf_kernel(const float* __restrict__ src, bf16* __restrict__ dst, size_t n)
{
  const size_t stride = (size_t)gridDim.x * blockDim.x;
  for (size_t i = (size_t)blockIdx.x * blockDim.x + threadIdx.x; i < n; i += stride)
    dst[i] = (bf16)src[i];
}

// ---------------------------------------------------------------------------
// Causal attention, one workgroup = (batch b, head h, 64-query tile).
// qkv: [4096, 768] bf16 rows = b*S+s, [q|k|v] each 256 wide, head stride 32.
// Scores & softmax in LDS fp32, P bf16 in LDS; both matmuls via WMMA.
// ---------------------------------------------------------------------------
__global__ void __launch_bounds__(256)
attention_kernel(const bf16* __restrict__ qkv, float* __restrict__ aout)
{
  constexpr int Sq = 512, Dm = 256, Dh = 32, LDQ = 768;
  constexpr int VST = 520, PST = 520;            // 1040B rows: 16B-aligned, bank-safe
  extern __shared__ char smem[];
  bf16*  Vt = (bf16*)smem;                               // [32][VST]  V transposed
  float* Sc = (float*)(smem + (size_t)32 * VST * 2);     // [64][512]  scores
  bf16*  P  = (bf16*)((char*)Sc + (size_t)64 * 512 * 4); // [64][PST]  probs

  const int q0 = blockIdx.x * 64;
  const int h  = blockIdx.y;
  const int b  = blockIdx.z;
  const int tid = threadIdx.x, lane = tid & 31, wid = tid >> 5;
  const size_t rowbase = (size_t)b * Sq;
  const int qc = h * Dh, kc = Dm + h * Dh, vc = 2 * Dm + h * Dh;

  // stage V^T : Vt[d][s] = V[s][d]  (b128 reads, scatter b16 stores)
  for (int c = tid; c < (Sq * Dh) / 8; c += 256) {
    const int s = c >> 2, dc = (c & 3) << 3;
    const u32x4 w = *reinterpret_cast<const u32x4*>(&qkv[(rowbase + s) * LDQ + vc + dc]);
    const bf16* wp = reinterpret_cast<const bf16*>(&w);
#pragma unroll
    for (int j = 0; j < 8; ++j) Vt[(dc + j) * VST + s] = wp[j];
  }
  __syncthreads();

  // scores: wave `wid` owns keys [wid*64, wid*64+64);  K-dim = Dh = 32 (1 WMMA)
  {
    const int key0 = wid * 64;
    FragA qf[4];
#pragma unroll
    for (int sm = 0; sm < 4; ++sm)
      load_fragA(qf[sm], qkv + (rowbase + q0) * LDQ + qc, LDQ, sm * 16, lane);
    const int rofs = (lane >> 4) * 8;
    const int nl   = lane & 15;
#pragma unroll
    for (int sn = 0; sn < 4; ++sn) {
      FragA kf;
      load_fragB(kf, qkv + rowbase * LDQ + kc, LDQ, key0 + sn * 16, lane);
      const int kg = key0 + sn * 16 + nl;
#pragma unroll
      for (int sm = 0; sm < 4; ++sm) {
        FragC acc;
#pragma unroll
        for (int r = 0; r < 8; ++r) acc.f[r] = 0.0f;
        acc.v = __builtin_amdgcn_wmma_f32_16x16x32_bf16(
            false, qf[sm].v, false, kf.v, (short)0, acc.v, false, false);
#pragma unroll
        for (int r = 0; r < 8; ++r) {
          const int q = sm * 16 + rofs + r;
          Sc[q * 512 + kg] = (kg <= q0 + q) ? acc.f[r] * 0.17677669529663689f : -1e10f;
        }
      }
    }
  }
  __syncthreads();

  // softmax: wave handles 8 rows, lane-strided over 512 keys
  for (int rr = 0; rr < 8; ++rr) {
    const int q = wid * 8 + rr;
    const float* srow = Sc + q * 512;
    float vals[16];
    float mx = -3.0e38f;
#pragma unroll
    for (int i = 0; i < 16; ++i) { vals[i] = srow[lane + i * 32]; mx = fmaxf(mx, vals[i]); }
#pragma unroll
    for (int off = 16; off; off >>= 1) mx = fmaxf(mx, __shfl_xor(mx, off, 32));
    float sum = 0.0f;
#pragma unroll
    for (int i = 0; i < 16; ++i) { vals[i] = __expf(vals[i] - mx); sum += vals[i]; }
#pragma unroll
    for (int off = 16; off; off >>= 1) sum += __shfl_xor(sum, off, 32);
    const float inv = 1.0f / sum;
#pragma unroll
    for (int i = 0; i < 16; ++i) P[q * PST + lane + i * 32] = (bf16)(vals[i] * inv);
  }
  __syncthreads();

  // P @ V : each wave one 16x16 output subtile, K-loop over 512 in steps of 32
  {
    const int sm = wid >> 1;          // 4 query sub-tiles
    const int sn = wid & 1;           // 2 dh sub-tiles
    FragC acc;
#pragma unroll
    for (int r = 0; r < 8; ++r) acc.f[r] = 0.0f;
    for (int k0 = 0; k0 < Sq; k0 += 32) {
      FragA pf, vf;
      load_fragA(pf, P + k0, PST, sm * 16, lane);
      load_fragB(vf, Vt + k0, VST, sn * 16, lane);
      acc.v = __builtin_amdgcn_wmma_f32_16x16x32_bf16(
          false, pf.v, false, vf.v, (short)0, acc.v, false, false);
    }
    const int rofs = (lane >> 4) * 8;
    const int col  = h * Dh + sn * 16 + (lane & 15);
#pragma unroll
    for (int r = 0; r < 8; ++r) {
      const int qg = q0 + sm * 16 + rofs + r;
      aout[(rowbase + qg) * Dm + col] = acc.f[r];
    }
  }
}

// ---------------------------------------------------------------------------
// Final global normalization: (x - mean) / std(ddof=1) + 1e-10  (deterministic)
// ---------------------------------------------------------------------------
__global__ void __launch_bounds__(256)
stat_partial_kernel(const float* __restrict__ x, size_t n,
                    float* __restrict__ ps, float* __restrict__ ps2)
{
  float s = 0.0f, s2 = 0.0f;
  const size_t stride = (size_t)gridDim.x * blockDim.x;
  for (size_t i = (size_t)blockIdx.x * blockDim.x + threadIdx.x; i < n; i += stride) {
    const float v = x[i]; s += v; s2 += v * v;
  }
#pragma unroll
  for (int off = 16; off; off >>= 1) { s += __shfl_xor(s, off, 32); s2 += __shfl_xor(s2, off, 32); }
  __shared__ float rs[8], rs2[8];
  const int tid = threadIdx.x;
  if ((tid & 31) == 0) { rs[tid >> 5] = s; rs2[tid >> 5] = s2; }
  __syncthreads();
  if (tid == 0) {
    float a = 0.0f, a2 = 0.0f;
    for (int i = 0; i < 8; ++i) { a += rs[i]; a2 += rs2[i]; }
    ps[blockIdx.x] = a; ps2[blockIdx.x] = a2;
  }
}

__global__ void __launch_bounds__(256)
stat_final_kernel(const float* __restrict__ ps, const float* __restrict__ ps2, int nb,
                  float* __restrict__ stats, float n)
{
  const int tid = threadIdx.x;
  float s = 0.0f, s2 = 0.0f;
  for (int i = tid; i < nb; i += 256) { s += ps[i]; s2 += ps2[i]; }
#pragma unroll
  for (int off = 16; off; off >>= 1) { s += __shfl_xor(s, off, 32); s2 += __shfl_xor(s2, off, 32); }
  __shared__ float rs[8], rs2[8];
  if ((tid & 31) == 0) { rs[tid >> 5] = s; rs2[tid >> 5] = s2; }
  __syncthreads();
  if (tid == 0) {
    float a = 0.0f, a2 = 0.0f;
    for (int i = 0; i < 8; ++i) { a += rs[i]; a2 += rs2[i]; }
    const float mean = a / n;
    const float var  = (a2 - n * mean * mean) / (n - 1.0f);
    stats[0] = mean;
    stats[1] = rsqrtf(var);
  }
}

__global__ void norm_kernel(const float* __restrict__ x, const float* __restrict__ stats,
                            float* __restrict__ out, size_t n)
{
  const float mean = stats[0], inv = stats[1];
  const size_t stride = (size_t)gridDim.x * blockDim.x;
  for (size_t i = (size_t)blockIdx.x * blockDim.x + threadIdx.x; i < n; i += stride)
    out[i] = (x[i] - mean) * inv + 1e-10f;
}

// ---------------------------------------------------------------------------
// Host orchestration
// ---------------------------------------------------------------------------
extern "C" void kernel_launch(void* const* d_in, const int* in_sizes, int n_in,
                              void* d_out, int out_size, void* d_ws, size_t ws_size,
                              hipStream_t stream)
{
  (void)in_sizes; (void)n_in; (void)out_size; (void)ws_size;

  const float* state = (const float*)d_in[0];
  const float* fc1_w = (const float*)d_in[1];  const float* fc1_b = (const float*)d_in[2];
  const float* fc2_w = (const float*)d_in[3];  const float* fc2_b = (const float*)d_in[4];
  const float* fc3_w = (const float*)d_in[5];  const float* fc3_b = (const float*)d_in[6];
  const float* fc4_w = (const float*)d_in[7];  const float* fc4_b = (const float*)d_in[8];
  const float* fc5_w = (const float*)d_in[9];  const float* fc5_b = (const float*)d_in[10];
  const float* pre_w = (const float*)d_in[11]; const float* pre_b = (const float*)d_in[12];
  const float* pos_w = (const float*)d_in[13];
  const float* enc_w = (const float*)d_in[14]; const float* enc_b = (const float*)d_in[15];
  const float* ln1_g = (const float*)d_in[16]; const float* ln1_b = (const float*)d_in[17];
  const float* ln2_g = (const float*)d_in[18]; const float* ln2_b = (const float*)d_in[19];
  const float* res_w1 = (const float*)d_in[20]; const float* res_b1 = (const float*)d_in[21];
  const float* res_w2 = (const float*)d_in[22]; const float* res_b2 = (const float*)d_in[23];
  const float* out_w  = (const float*)d_in[24]; const float* out_b  = (const float*)d_in[25];

  const int M = 4096;                  // B*S
  size_t off = 0;
  auto alloc = [&](size_t bytes) -> void* {
    void* p = (char*)d_ws + off;
    off += (bytes + 255) & ~(size_t)255;
    return p;
  };

  bf16* wb_fc1 = (bf16*)alloc((size_t)4096 * 4096 * 2);
  bf16* wb_fc2 = (bf16*)alloc((size_t)4096 * 2048 * 2);
  bf16* wb_fc3 = (bf16*)alloc((size_t)2048 * 1024 * 2);
  bf16* wb_fc4 = (bf16*)alloc((size_t)1024 * 512 * 2);
  bf16* wb_fc5 = (bf16*)alloc((size_t)512 * 256 * 2);
  bf16* wb_pre = (bf16*)alloc((size_t)256 * 256 * 2);
  bf16* wb_enc = (bf16*)alloc((size_t)8 * 256 * 768 * 2);
  bf16* wb_r1  = (bf16*)alloc((size_t)8 * 256 * 1024 * 2);
  bf16* wb_r2  = (bf16*)alloc((size_t)8 * 1024 * 256 * 2);
  bf16* wb_out = (bf16*)alloc((size_t)256 * 128 * 2);
  bf16* act0   = (bf16*)alloc((size_t)M * 4096 * 2);
  bf16* act1   = (bf16*)alloc((size_t)M * 4096 * 2);
  float* xf    = (float*)alloc((size_t)M * 256 * 4);
  bf16* xnbf   = (bf16*)alloc((size_t)M * 256 * 2);
  bf16* qkvb   = (bf16*)alloc((size_t)M * 768 * 2);
  float* aout  = (float*)alloc((size_t)M * 256 * 4);
  bf16* hbf    = (bf16*)alloc((size_t)M * 1024 * 2);
  float* h2f   = (float*)alloc((size_t)M * 256 * 4);
  float* encf  = (float*)alloc((size_t)M * 128 * 4);
  float* ps    = (float*)alloc(512 * 4);
  float* ps2   = (float*)alloc(512 * 4);
  float* stats = (float*)alloc(2 * 4);

  auto cvt = [&](const float* s, bf16* d, size_t n) {
    f2bf_kernel<<<512, 256, 0, stream>>>(s, d, n);
  };
  cvt(fc1_w, wb_fc1, (size_t)4096 * 4096);
  cvt(fc2_w, wb_fc2, (size_t)4096 * 2048);
  cvt(fc3_w, wb_fc3, (size_t)2048 * 1024);
  cvt(fc4_w, wb_fc4, (size_t)1024 * 512);
  cvt(fc5_w, wb_fc5, (size_t)512 * 256);
  cvt(pre_w, wb_pre, (size_t)256 * 256);
  cvt(enc_w, wb_enc, (size_t)8 * 256 * 768);
  cvt(res_w1, wb_r1, (size_t)8 * 256 * 1024);
  cvt(res_w2, wb_r2, (size_t)8 * 1024 * 256);
  cvt(out_w, wb_out, (size_t)256 * 128);
  cvt(state, act0, (size_t)M * 4096);      // reshape is a no-op

  const dim3 blk(NT);
  // MLP trunk (relu x4, tanh)
  gemm_bf16_kernel<1, false><<<dim3(32, 32), blk, 0, stream>>>(act0, wb_fc1, fc1_b, nullptr, act1, nullptr, M, 4096, 4096);
  gemm_bf16_kernel<1, false><<<dim3(16, 32), blk, 0, stream>>>(act1, wb_fc2, fc2_b, nullptr, act0, nullptr, M, 2048, 4096);
  gemm_bf16_kernel<1, false><<<dim3(8, 32),  blk, 0, stream>>>(act0, wb_fc3, fc3_b, nullptr, act1, nullptr, M, 1024, 2048);
  gemm_bf16_kernel<1, false><<<dim3(4, 32),  blk, 0, stream>>>(act1, wb_fc4, fc4_b, nullptr, act0, nullptr, M, 512, 1024);
  gemm_bf16_kernel<2, false><<<dim3(2, 32),  blk, 0, stream>>>(act0, wb_fc5, fc5_b, nullptr, act1, nullptr, M, 256, 512);
  // pre-projection + positional add -> fp32 master x
  gemm_bf16_kernel<0, true><<<dim3(2, 32),   blk, 0, stream>>>(act1, wb_pre, pre_b, pos_w, nullptr, xf, M, 256, 256);

  const size_t smem_attn = (size_t)32 * 520 * 2 + (size_t)64 * 512 * 4 + (size_t)64 * 520 * 2;
  for (int l = 0; l < 8; ++l) {
    layernorm_kernel<<<4096, 256, 0, stream>>>(xf, ln1_g + l * 256, ln1_b + l * 256, xnbf);
    gemm_bf16_kernel<0, false><<<dim3(6, 32), blk, 0, stream>>>(
        xnbf, wb_enc + (size_t)l * 256 * 768, enc_b + l * 768, nullptr, qkvb, nullptr, M, 768, 256);
    attention_kernel<<<dim3(8, 8, 8), 256, smem_attn, stream>>>(qkvb, aout);
    add_layernorm_kernel<<<4096, 256, 0, stream>>>(xf, aout, ln2_g + l * 256, ln2_b + l * 256, xnbf);
    gemm_bf16_kernel<3, false><<<dim3(8, 32), blk, 0, stream>>>(
        xnbf, wb_r1 + (size_t)l * 256 * 1024, res_b1 + l * 1024, nullptr, hbf, nullptr, M, 1024, 256);
    gemm_bf16_kernel<0, false><<<dim3(2, 32), blk, 0, stream>>>(
        hbf, wb_r2 + (size_t)l * 1024 * 256, res_b2 + l * 256, nullptr, nullptr, h2f, M, 256, 1024);
    add_bf_kernel<<<512, 256, 0, stream>>>(xf, h2f, xnbf, (size_t)M * 256);
  }

  // output projection + global normalization
  gemm_bf16_kernel<0, false><<<dim3(1, 32), blk, 0, stream>>>(
      xnbf, wb_out, out_b, nullptr, nullptr, encf, M, 128, 256);
  stat_partial_kernel<<<512, 256, 0, stream>>>(encf, (size_t)M * 128, ps, ps2);
  stat_final_kernel<<<1, 256, 0, stream>>>(ps, ps2, 512, stats, (float)((size_t)M * 128));
  norm_kernel<<<512, 256, 0, stream>>>(encf, stats, (float*)d_out, (size_t)M * 128);
}
